// Sg2ImModel_7739531067394
// MI455X (gfx1250) — compile-verified, matrix operand order checked by reading, not run on hardware
//
#include <hip/hip_runtime.h>
#include <stdint.h>

#define O_CNT 16000
#define T_CNT 50000
#define MT_PAD 50016          // T padded to a multiple of 32 (1563 M-blocks)
#define DEMB  64
#define DG    128
#define HDIM  512
#define NOUT  (2*HDIM + DG)   // 1152

typedef __attribute__((ext_vector_type(16))) __bf16        v16bf;
typedef __attribute__((ext_vector_type(16))) unsigned short v16u;
typedef __attribute__((ext_vector_type(8)))  unsigned short v8u;
typedef __attribute__((ext_vector_type(8)))  float          v8f;

static __device__ __forceinline__ unsigned short f2bf(float f) {
  unsigned int u = __float_as_uint(f);
  u += 0x7FFFu + ((u >> 16) & 1u);          // round-to-nearest-even
  return (unsigned short)(u >> 16);
}

// K index for element e (0..15) of a 16-bit A/B fragment (CDNA5 ISA 7.12.2)
static __device__ __forceinline__ int kmap(int e, int half) {
  int pair = e >> 1, sub = e & 1;
  int base = (pair < 4) ? (pair * 2) : (16 + (pair - 4) * 2);
  return base + sub + half * 8;
}

static __device__ __forceinline__ v16bf load_frag_a(const unsigned short* base) {
  v8u lo = *(const v8u*)(base);        // K-run 0..7 (this half)
  v8u hi = *(const v8u*)(base + 16);   // K-run 16..23 (this half)
  v16u t;
#pragma unroll
  for (int i = 0; i < 8; i++) { t[i] = lo[i]; t[8 + i] = hi[i]; }
  return __builtin_bit_cast(v16bf, t);
}

static __device__ __forceinline__ v16bf load_frag_b(const unsigned short* base) {
  v8u lo = *(const v8u*)(base);        // packed fragment: 16 contiguous bf16/lane
  v8u hi = *(const v8u*)(base + 8);
  v16u t;
#pragma unroll
  for (int i = 0; i < 8; i++) { t[i] = lo[i]; t[8 + i] = hi[i]; }
  return __builtin_bit_cast(v16bf, t);
}

static __device__ __forceinline__ v8f wmma_bf16(v16bf a, v16bf b, v8f c) {
  return __builtin_amdgcn_wmma_f32_16x16x32_bf16(false, a, false, b, (short)0, c,
                                                 false, false);
}

// ---- weight packing: f32 (K x N row-major) -> fragment-major bf16 -------------
__global__ void pack_w_kernel(const float* __restrict__ W, unsigned short* __restrict__ out,
                              int K, int N) {
  int i = blockIdx.x * blockDim.x + threadIdx.x;
  int total = K * N;
  if (i >= total) return;
  int nTiles = N >> 4;
  int f = i >> 9;            // 512 elems per 32x16 fragment
  int w = i & 511;
  int lane = w >> 4;
  int e    = w & 15;
  int kTile = f / nTiles, nTile = f % nTiles;
  int half = lane >> 4;
  int n = nTile * 16 + (lane & 15);
  int k = kTile * 32 + kmap(e, half);
  out[i] = f2bf(W[(size_t)k * N + n]);
}

// ---- gathers ------------------------------------------------------------------
__global__ void gather0_kernel(const int* __restrict__ objs, const int* __restrict__ triples,
                               const float* __restrict__ obj_emb, const float* __restrict__ pred_emb,
                               unsigned short* __restrict__ curt) {
  const int W = 3 * DEMB;
  int i = blockIdx.x * blockDim.x + threadIdx.x;
  if (i >= T_CNT * W) return;
  int t = i / W, c = i % W;
  float v;
  if (c < DEMB) {
    int s = triples[t * 3 + 0];
    v = obj_emb[(size_t)objs[s] * DEMB + c];
  } else if (c < 2 * DEMB) {
    int p = triples[t * 3 + 1];
    v = pred_emb[(size_t)p * DEMB + (c - DEMB)];
  } else {
    int o = triples[t * 3 + 2];
    v = obj_emb[(size_t)objs[o] * DEMB + (c - 2 * DEMB)];
  }
  curt[(size_t)t * W + c] = f2bf(v);
}

__global__ void gatherN_kernel(const int* __restrict__ triples, const float* __restrict__ objv,
                               const float* __restrict__ predv, unsigned short* __restrict__ curt) {
  const int W = 3 * DG;
  int i = blockIdx.x * blockDim.x + threadIdx.x;
  if (i >= T_CNT * W) return;
  int t = i / W, c = i % W;
  float v;
  if (c < DG)            v = objv[(size_t)triples[t * 3 + 0] * DG + c];
  else if (c < 2 * DG)   v = predv[(size_t)t * DG + (c - DG)];
  else                   v = objv[(size_t)triples[t * 3 + 2] * DG + (c - 2 * DG)];
  curt[(size_t)t * W + c] = f2bf(v);
}

// ---- misc elementwise ---------------------------------------------------------
__global__ void zero_f32_kernel(float* __restrict__ p, int n) {
  int i = blockIdx.x * blockDim.x + threadIdx.x;
  if (i < n) p[i] = 0.0f;
}

__global__ void count_kernel(const int* __restrict__ triples, float* __restrict__ cnts) {
  int t = blockIdx.x * blockDim.x + threadIdx.x;
  if (t >= T_CNT) return;
  atomicAdd(&cnts[triples[t * 3 + 0]], 1.0f);
  atomicAdd(&cnts[triples[t * 3 + 2]], 1.0f);
}

__global__ void invc_kernel(const float* __restrict__ cnts, float* __restrict__ invc) {
  int i = blockIdx.x * blockDim.x + threadIdx.x;
  if (i < O_CNT) invc[i] = 1.0f / fmaxf(cnts[i], 1.0f);
}

__global__ void div_bf16_kernel(const float* __restrict__ pooled, const float* __restrict__ invc,
                                unsigned short* __restrict__ pb) {
  int i = blockIdx.x * blockDim.x + threadIdx.x;
  if (i < O_CNT * HDIM) pb[i] = f2bf(pooled[i] * invc[i >> 9]);
}

// ---- bf16 WMMA GEMM: 32x64 per wave, software-pipelined -----------------------
// 8 WMMAs per K-step on double-buffered fragments: next iteration's 2 A + 4 B
// fragments are loaded before the current 8 WMMAs issue, so each wait covers
// loads issued a full MMA-group earlier.
template <bool RELU, bool OUT_BF16>
__global__ void gemm_kernel(const unsigned short* __restrict__ A,
                            const unsigned short* __restrict__ Bp,
                            const float* __restrict__ bias, void* __restrict__ out,
                            int mBlocks, int K, int N) {
  int wave  = threadIdx.x >> 5;
  int tile  = blockIdx.x * (blockDim.x >> 5) + wave;
  int nStrips = N >> 6, nTiles = N >> 4;
  if (tile >= mBlocks * nStrips) return;          // wave-uniform exit: EXEC all-1
  int mBlk = tile / nStrips, strip = tile % nStrips;
  int lane = threadIdx.x & 31;
  int half = lane >> 4, lr = lane & 15;

  v8f acc[8] = {};                                // [m*4 + j]
  const unsigned short* aRow0 = A + (size_t)(mBlk * 32 + lr) * K + half * 8;
  const unsigned short* aRow1 = aRow0 + (size_t)16 * K;
  const unsigned short* bBase = Bp + (size_t)(strip * 4) * 512 + lane * 16;
  int kTiles = K >> 5;

  v16bf a0 = load_frag_a(aRow0);
  v16bf a1 = load_frag_a(aRow1);
  v16bf b0 = load_frag_b(bBase);
  v16bf b1 = load_frag_b(bBase + 512);
  v16bf b2 = load_frag_b(bBase + 1024);
  v16bf b3 = load_frag_b(bBase + 1536);

  for (int kt = 0; kt < kTiles - 1; kt++) {
    v16bf a0n = load_frag_a(aRow0 + (kt + 1) * 32);
    v16bf a1n = load_frag_a(aRow1 + (kt + 1) * 32);
    const unsigned short* bk = bBase + (size_t)(kt + 1) * nTiles * 512;
    v16bf b0n = load_frag_b(bk);
    v16bf b1n = load_frag_b(bk + 512);
    v16bf b2n = load_frag_b(bk + 1024);
    v16bf b3n = load_frag_b(bk + 1536);
    acc[0] = wmma_bf16(a0, b0, acc[0]);
    acc[1] = wmma_bf16(a0, b1, acc[1]);
    acc[2] = wmma_bf16(a0, b2, acc[2]);
    acc[3] = wmma_bf16(a0, b3, acc[3]);
    acc[4] = wmma_bf16(a1, b0, acc[4]);
    acc[5] = wmma_bf16(a1, b1, acc[5]);
    acc[6] = wmma_bf16(a1, b2, acc[6]);
    acc[7] = wmma_bf16(a1, b3, acc[7]);
    a0 = a0n; a1 = a1n; b0 = b0n; b1 = b1n; b2 = b2n; b3 = b3n;
  }
  acc[0] = wmma_bf16(a0, b0, acc[0]);
  acc[1] = wmma_bf16(a0, b1, acc[1]);
  acc[2] = wmma_bf16(a0, b2, acc[2]);
  acc[3] = wmma_bf16(a0, b3, acc[3]);
  acc[4] = wmma_bf16(a1, b0, acc[4]);
  acc[5] = wmma_bf16(a1, b1, acc[5]);
  acc[6] = wmma_bf16(a1, b2, acc[6]);
  acc[7] = wmma_bf16(a1, b3, acc[7]);

#pragma unroll
  for (int m = 0; m < 2; m++) {
#pragma unroll
    for (int j = 0; j < 4; j++) {
      int col = (strip * 4 + j) * 16 + lr;
      float bv = bias[col];
#pragma unroll
      for (int v = 0; v < 8; v++) {
        float val = acc[m * 4 + j][v] + bv;
        if (RELU) val = fmaxf(val, 0.0f);
        size_t idx = (size_t)(mBlk * 32 + m * 16 + half * 8 + v) * N + col;
        if (OUT_BF16) ((unsigned short*)out)[idx] = f2bf(val);
        else          ((float*)out)[idx] = val;
      }
    }
  }
}

// ---- GEMM2 fused with scatter epilogue (avoids materializing T x 1152) --------
// Column strips of 64: strips 0-7 -> new_s, strips 8-9 -> new_p,
// strips 10-17 -> new_o. Boundaries (512, 640) divide by 64 -> uniform branch.
// Rows >= T_CNT (padding) are skipped in the epilogue only.
__global__ void gemm2_scatter_kernel(const unsigned short* __restrict__ A,
                                     const unsigned short* __restrict__ Bp,
                                     const float* __restrict__ bias,
                                     const int* __restrict__ triples,
                                     float* __restrict__ pooled, float* __restrict__ predv,
                                     int mBlocks) {
  const int K = HDIM, N = NOUT;
  int wave  = threadIdx.x >> 5;
  int tile  = blockIdx.x * (blockDim.x >> 5) + wave;
  int nStrips = N >> 6, nTiles = N >> 4;          // 18, 72
  if (tile >= mBlocks * nStrips) return;
  int mBlk = tile / nStrips, strip = tile % nStrips;
  int lane = threadIdx.x & 31;
  int half = lane >> 4, lr = lane & 15;

  v8f acc[8] = {};
  const unsigned short* aRow0 = A + (size_t)(mBlk * 32 + lr) * K + half * 8;
  const unsigned short* aRow1 = aRow0 + (size_t)16 * K;
  const unsigned short* bBase = Bp + (size_t)(strip * 4) * 512 + lane * 16;
  int kTiles = K >> 5;

  v16bf a0 = load_frag_a(aRow0);
  v16bf a1 = load_frag_a(aRow1);
  v16bf b0 = load_frag_b(bBase);
  v16bf b1 = load_frag_b(bBase + 512);
  v16bf b2 = load_frag_b(bBase + 1024);
  v16bf b3 = load_frag_b(bBase + 1536);

  for (int kt = 0; kt < kTiles - 1; kt++) {
    v16bf a0n = load_frag_a(aRow0 + (kt + 1) * 32);
    v16bf a1n = load_frag_a(aRow1 + (kt + 1) * 32);
    const unsigned short* bk = bBase + (size_t)(kt + 1) * nTiles * 512;
    v16bf b0n = load_frag_b(bk);
    v16bf b1n = load_frag_b(bk + 512);
    v16bf b2n = load_frag_b(bk + 1024);
    v16bf b3n = load_frag_b(bk + 1536);
    acc[0] = wmma_bf16(a0, b0, acc[0]);
    acc[1] = wmma_bf16(a0, b1, acc[1]);
    acc[2] = wmma_bf16(a0, b2, acc[2]);
    acc[3] = wmma_bf16(a0, b3, acc[3]);
    acc[4] = wmma_bf16(a1, b0, acc[4]);
    acc[5] = wmma_bf16(a1, b1, acc[5]);
    acc[6] = wmma_bf16(a1, b2, acc[6]);
    acc[7] = wmma_bf16(a1, b3, acc[7]);
    a0 = a0n; a1 = a1n; b0 = b0n; b1 = b1n; b2 = b2n; b3 = b3n;
  }
  acc[0] = wmma_bf16(a0, b0, acc[0]);
  acc[1] = wmma_bf16(a0, b1, acc[1]);
  acc[2] = wmma_bf16(a0, b2, acc[2]);
  acc[3] = wmma_bf16(a0, b3, acc[3]);
  acc[4] = wmma_bf16(a1, b0, acc[4]);
  acc[5] = wmma_bf16(a1, b1, acc[5]);
  acc[6] = wmma_bf16(a1, b2, acc[6]);
  acc[7] = wmma_bf16(a1, b3, acc[7]);

  float bv[4];
#pragma unroll
  for (int j = 0; j < 4; j++) bv[j] = bias[(strip * 4 + j) * 16 + lr];
  int colBase = strip * 64 + lr;

  if (strip < 8) {                                // new_s -> pooled[s_idx]
#pragma unroll
    for (int m = 0; m < 2; m++) {
#pragma unroll
      for (int v = 0; v < 8; v++) {
        int r = mBlk * 32 + m * 16 + half * 8 + v;
        if (r < T_CNT) {
          int s = triples[r * 3 + 0];
          float* dst = &pooled[(size_t)s * HDIM + colBase];
#pragma unroll
          for (int j = 0; j < 4; j++) atomicAdd(dst + j * 16, acc[m * 4 + j][v] + bv[j]);
        }
      }
    }
  } else if (strip < 10) {                        // new_p -> pred_vecs
#pragma unroll
    for (int m = 0; m < 2; m++) {
#pragma unroll
      for (int v = 0; v < 8; v++) {
        int r = mBlk * 32 + m * 16 + half * 8 + v;
        if (r < T_CNT) {
          float* dst = &predv[(size_t)r * DG + (colBase - HDIM)];
#pragma unroll
          for (int j = 0; j < 4; j++) dst[j * 16] = acc[m * 4 + j][v] + bv[j];
        }
      }
    }
  } else {                                        // new_o -> pooled[o_idx]
#pragma unroll
    for (int m = 0; m < 2; m++) {
#pragma unroll
      for (int v = 0; v < 8; v++) {
        int r = mBlk * 32 + m * 16 + half * 8 + v;
        if (r < T_CNT) {
          int o = triples[r * 3 + 2];
          float* dst = &pooled[(size_t)o * HDIM + (colBase - HDIM - DG)];
#pragma unroll
          for (int j = 0; j < 4; j++) atomicAdd(dst + j * 16, acc[m * 4 + j][v] + bv[j]);
        }
      }
    }
  }
}

// ---- final column mean --------------------------------------------------------
__global__ void mean_kernel(const float* __restrict__ objv, float* __restrict__ out) {
  __shared__ float red[256];
  int col = blockIdx.x;
  float s = 0.0f;
  for (int r = threadIdx.x; r < O_CNT; r += blockDim.x) s += objv[(size_t)r * DG + col];
  red[threadIdx.x] = s;
  __syncthreads();
  for (int st = 128; st > 0; st >>= 1) {
    if ((int)threadIdx.x < st) red[threadIdx.x] += red[threadIdx.x + st];
    __syncthreads();
  }
  if (threadIdx.x == 0) out[col] = red[0] / (float)O_CNT;
}

extern "C" void kernel_launch(void* const* d_in, const int* in_sizes, int n_in,
                              void* d_out, int out_size, void* d_ws, size_t ws_size,
                              hipStream_t stream) {
  (void)in_sizes; (void)n_in; (void)out_size; (void)ws_size;
  const int*   objs    = (const int*)d_in[0];
  const int*   triples = (const int*)d_in[1];
  const float* obj_emb = (const float*)d_in[2];
  const float* pred_emb= (const float*)d_in[3];

  char* ws = (char*)d_ws;
  size_t off = 0;
  auto alloc = [&](size_t bytes) -> char* {
    char* p = ws + off;
    off += (bytes + 255) & ~(size_t)255;
    return p;
  };

  unsigned short *pw1[5], *pw2[5], *pw3[5], *pw4[5];
  pw1[0] = (unsigned short*)alloc((size_t)192 * 512 * 2);
  for (int i = 1; i < 5; i++) pw1[i] = (unsigned short*)alloc((size_t)384 * 512 * 2);
  for (int i = 0; i < 5; i++) pw2[i] = (unsigned short*)alloc((size_t)512 * 1152 * 2);
  for (int i = 0; i < 5; i++) pw3[i] = (unsigned short*)alloc((size_t)512 * 512 * 2);
  for (int i = 0; i < 5; i++) pw4[i] = (unsigned short*)alloc((size_t)512 * 128 * 2);
  unsigned short* curt    = (unsigned short*)alloc((size_t)MT_PAD * 384 * 2);
  unsigned short* hbuf    = (unsigned short*)alloc((size_t)MT_PAD * 512 * 2);
  float*          pooled  = (float*)alloc((size_t)O_CNT * 512 * 4);
  unsigned short* pooledb = (unsigned short*)alloc((size_t)O_CNT * 512 * 2);
  unsigned short* h2      = (unsigned short*)alloc((size_t)O_CNT * 512 * 2);
  float*          objv    = (float*)alloc((size_t)O_CNT * 128 * 4);
  float*          predv   = (float*)alloc((size_t)T_CNT * 128 * 4);
  float*          cnts    = (float*)alloc((size_t)O_CNT * 4);
  float*          invc    = (float*)alloc((size_t)O_CNT * 4);

  const int TB = 256;
  auto cdiv = [](long long a, long long b) { return (int)((a + b - 1) / b); };
  auto pack = [&](const float* W, unsigned short* dst, int K, int N) {
    pack_w_kernel<<<cdiv((long long)K * N, TB), TB, 0, stream>>>(W, dst, K, N);
  };

  pack((const float*)d_in[4],  pw1[0], 192, 512);
  pack((const float*)d_in[6],  pw2[0], 512, 1152);
  pack((const float*)d_in[8],  pw3[0], 512, 512);
  pack((const float*)d_in[10], pw4[0], 512, 128);
  for (int j = 0; j < 4; j++) {
    pack((const float*)d_in[12] + (size_t)j * 384 * 512,  pw1[1 + j], 384, 512);
    pack((const float*)d_in[14] + (size_t)j * 512 * 1152, pw2[1 + j], 512, 1152);
    pack((const float*)d_in[16] + (size_t)j * 512 * 512,  pw3[1 + j], 512, 512);
    pack((const float*)d_in[18] + (size_t)j * 512 * 128,  pw4[1 + j], 512, 128);
  }

  // degree counts (layer-invariant)
  zero_f32_kernel<<<cdiv(O_CNT, TB), TB, 0, stream>>>(cnts, O_CNT);
  count_kernel<<<cdiv(T_CNT, TB), TB, 0, stream>>>(triples, cnts);
  invc_kernel<<<cdiv(O_CNT, TB), TB, 0, stream>>>(cnts, invc);

  const int mbT = MT_PAD / 32;   // 1563
  const int mbO = O_CNT / 32;    // 500

  for (int layer = 0; layer < 5; layer++) {
    const float *b1, *b2, *b3, *b4;
    int Kin;
    if (layer == 0) {
      b1 = (const float*)d_in[5];  b2 = (const float*)d_in[7];
      b3 = (const float*)d_in[9];  b4 = (const float*)d_in[11];
      Kin = 3 * DEMB;
      gather0_kernel<<<cdiv((long long)T_CNT * 192, TB), TB, 0, stream>>>(
          objs, triples, obj_emb, pred_emb, curt);
    } else {
      int j = layer - 1;
      b1 = (const float*)d_in[13] + (size_t)j * 512;
      b2 = (const float*)d_in[15] + (size_t)j * 1152;
      b3 = (const float*)d_in[17] + (size_t)j * 512;
      b4 = (const float*)d_in[19] + (size_t)j * 128;
      Kin = 3 * DG;
      gatherN_kernel<<<cdiv((long long)T_CNT * 384, TB), TB, 0, stream>>>(
          triples, objv, predv, curt);
    }

    gemm_kernel<true, true><<<cdiv((long long)mbT * (HDIM / 64), TB / 32), TB, 0, stream>>>(
        curt, pw1[layer], b1, hbuf, mbT, Kin, HDIM);

    zero_f32_kernel<<<cdiv((long long)O_CNT * HDIM, TB), TB, 0, stream>>>(pooled, O_CNT * HDIM);

    gemm2_scatter_kernel<<<cdiv((long long)mbT * (NOUT / 64), TB / 32), TB, 0, stream>>>(
        hbuf, pw2[layer], b2, triples, pooled, predv, mbT);

    div_bf16_kernel<<<cdiv((long long)O_CNT * HDIM, TB), TB, 0, stream>>>(pooled, invc, pooledb);

    gemm_kernel<true, true><<<cdiv((long long)mbO * (HDIM / 64), TB / 32), TB, 0, stream>>>(
        pooledb, pw3[layer], b3, h2, mbO, HDIM, HDIM);

    gemm_kernel<false, false><<<cdiv((long long)mbO * (DG / 64), TB / 32), TB, 0, stream>>>(
        h2, pw4[layer], b4, objv, mbO, HDIM, DG);
  }

  mean_kernel<<<DG, 256, 0, stream>>>(objv, (float*)d_out);
}
